// AttentionAggregator_47364899340956
// MI455X (gfx1250) — matrix operand order repaired and checked by reading, not compiled, not run on hardware
//
#include <hip/hip_runtime.h>

#define LEAKY_ALPHA 0.2f

typedef __attribute__((ext_vector_type(16))) __bf16 v16bf;
typedef __attribute__((ext_vector_type(8)))  __bf16 v8bf;
typedef __attribute__((ext_vector_type(8)))  float  v8f;

__device__ __forceinline__ float leaky(float x) { return x >= 0.0f ? x : LEAKY_ALPHA * x; }

// ---------------------------------------------------------------------------
// Kernel 0: wa[j] = dot(W[j,:], a)  and  Wt_{hi,lo}[j][k] = split_bf16(W[k][j])
// grid = 256 blocks (one per output row j), 256 threads.
// ---------------------------------------------------------------------------
__global__ void __launch_bounds__(256)
prep_kernel(const float* __restrict__ W,      // (256,256) row-major (Din,Dout)
            const float* __restrict__ a,      // (256)
            float*  __restrict__ wa,          // (256)
            __bf16* __restrict__ Wt_hi,       // (256,256) = W^T
            __bf16* __restrict__ Wt_lo) {
    __shared__ float red[256];
    const int j = blockIdx.x;
    const int t = threadIdx.x;

    // wa[j] = sum_e W[j][e] * a[e]   (coalesced row read)
    red[t] = W[j * 256 + t] * a[t];
    __syncthreads();
    for (int s = 128; s > 0; s >>= 1) {
        if (t < s) red[t] += red[t + s];
        __syncthreads();
    }
    if (t == 0) wa[j] = red[0];

    // Wt[j][t] = W[t][j], split into bf16 hi + lo (bf16x3 emulation of f32)
    float x = W[t * 256 + j];
    __bf16 hi = (__bf16)x;
    __bf16 lo = (__bf16)(x - (float)hi);
    Wt_hi[j * 256 + t] = hi;
    Wt_lo[j * 256 + t] = lo;
}

// ---------------------------------------------------------------------------
// Kernel 1: per batch row b — logits vs wa, leaky+softmax over 32 neighbors,
// s[b,:] = self[b,:] + sum_n coef_n * neigh[b,n,:]; store bf16 hi/lo split.
// grid = B blocks, 256 threads. One HBM pass over neigh_vecs (512 MB).
// ---------------------------------------------------------------------------
__global__ void __launch_bounds__(256)
attn_kernel(const float* __restrict__ selfv_g,   // (B,256)
            const float* __restrict__ neigh,     // (B,32,256)
            const float* __restrict__ wa_g,      // (256)
            __bf16* __restrict__ s_hi,           // (B,256)
            __bf16* __restrict__ s_lo) {
    __shared__ float tile[32 * 256];   // 32 KB neighbor tile
    __shared__ float lwa[256];
    __shared__ float llog[32];
    __shared__ float lcoef[32];
    __shared__ float sred[8];
    __shared__ float slog;

    const int b   = blockIdx.x;
    const int tid = threadIdx.x;

    lwa[tid] = wa_g[tid];
    // Coalesced float4 tile load: 2048 float4 over 256 threads
    const float4* src = (const float4*)(neigh + (size_t)b * 32 * 256);
    float4* dst = (float4*)tile;
#pragma unroll
    for (int i = 0; i < 8; ++i) dst[i * 256 + tid] = src[i * 256 + tid];
    const float sv = selfv_g[(size_t)b * 256 + tid];
    __syncthreads();

    // Neighbor logits: 8 threads per neighbor (contiguous lanes within a wave)
    {
        const int n = tid >> 3;
        const int t = tid & 7;
        float p = 0.0f;
        const float* row = tile + n * 256 + t * 32;
        const float* w   = lwa + t * 32;
#pragma unroll
        for (int e = 0; e < 32; ++e) p += row[e] * w[e];
        p += __shfl_down(p, 4, 8);
        p += __shfl_down(p, 2, 8);
        p += __shfl_down(p, 1, 8);
        if (t == 0) llog[n] = p;
    }
    // Self logit: 256-wide reduction (wave32 shuffles + LDS)
    {
        float p = sv * lwa[tid];
#pragma unroll
        for (int off = 16; off > 0; off >>= 1) p += __shfl_down(p, off, 32);
        if ((tid & 31) == 0) sred[tid >> 5] = p;
    }
    __syncthreads();
    if (tid == 0) {
        float s = 0.0f;
        for (int i = 0; i < 8; ++i) s += sred[i];
        slog = s;
    }
    __syncthreads();

    // Softmax over 32 neighbors in wave 0 (warpSize == 32 on gfx1250)
    if (tid < 32) {
        float l  = leaky(llog[tid] + slog);
        float mx = l;
#pragma unroll
        for (int off = 16; off > 0; off >>= 1) mx = fmaxf(mx, __shfl_xor(mx, off, 32));
        float e = __expf(l - mx);
        float sum = e;
#pragma unroll
        for (int off = 16; off > 0; off >>= 1) sum += __shfl_xor(sum, off, 32);
        lcoef[tid] = e / sum;
    }
    __syncthreads();

    // s[b,d] = self[b,d] + sum_n coef[n] * neigh[b,n,d]; bf16 hi/lo split
    float acc = sv;
#pragma unroll
    for (int n = 0; n < 32; ++n) acc += lcoef[n] * tile[n * 256 + tid];
    __bf16 hi = (__bf16)acc;
    __bf16 lo = (__bf16)(acc - (float)hi);
    s_hi[(size_t)b * 256 + tid] = hi;
    s_lo[(size_t)b * 256 + tid] = lo;
}

// ---------------------------------------------------------------------------
// Kernel 2: out = leaky( S @ W + bias ) via v_wmma_f32_16x16x32_bf16,
// bf16x3 split (hi*hi + hi*lo + lo*hi) for near-f32 accuracy.
// Block = 128 threads = 4 waves; each wave owns one 16x16 tile, K=256.
// grid = (N/16, M/64). A layout: 16-bit 16x32; B layout: 16-bit 32x16
// (lane half selects K sub-range per ISA 7.12.2).
// ---------------------------------------------------------------------------
__global__ void __launch_bounds__(128)
gemm_kernel(const __bf16* __restrict__ s_hi,    // (M,256)
            const __bf16* __restrict__ s_lo,
            const __bf16* __restrict__ Wt_hi,   // (256,256) = W^T (N-major)
            const __bf16* __restrict__ Wt_lo,
            const float*  __restrict__ bias,    // (256)
            float* __restrict__ out) {          // (M,256)
    const int lane  = threadIdx.x & 31;
    const int wave  = threadIdx.x >> 5;
    const int half  = lane >> 4;        // 0: lanes 0-15, 1: lanes 16-31
    const int l15   = lane & 15;
    const int mbase = (blockIdx.y * 4 + wave) * 16;
    const int nbase = blockIdx.x * 16;
    const int m = mbase + l15;          // A row for this lane
    const int n = nbase + l15;          // B column for this lane

    v8f acc = {};
#pragma unroll
    for (int kb = 0; kb < 256; kb += 32) {
        // A 16x32 bf16: VGPR0-3 = K[half*16 .. +8), VGPR4-7 = K[half*16+8 .. +8)
        v8bf a0h = *(const v8bf*)(s_hi + (size_t)m * 256 + kb + half * 16);
        v8bf a1h = *(const v8bf*)(s_hi + (size_t)m * 256 + kb + 8 + half * 16);
        v8bf a0l = *(const v8bf*)(s_lo + (size_t)m * 256 + kb + half * 16);
        v8bf a1l = *(const v8bf*)(s_lo + (size_t)m * 256 + kb + 8 + half * 16);
        v16bf ah, al;
#pragma unroll
        for (int i = 0; i < 8; ++i) {
            ah[i] = a0h[i]; ah[8 + i] = a1h[i];
            al[i] = a0l[i]; al[8 + i] = a1l[i];
        }
        // B 32x16 bf16: lanes 0-15 hold K[kb..kb+16), lanes 16-31 K[kb+16..kb+32)
        v16bf bh = *(const v16bf*)(Wt_hi + (size_t)n * 256 + kb + half * 16);
        v16bf bl = *(const v16bf*)(Wt_lo + (size_t)n * 256 + kb + half * 16);

        acc = __builtin_amdgcn_wmma_f32_16x16x32_bf16(false, al, false, bh, (short)0, acc, false, false);
        acc = __builtin_amdgcn_wmma_f32_16x16x32_bf16(false, ah, false, bl, (short)0, acc, false, false);
        acc = __builtin_amdgcn_wmma_f32_16x16x32_bf16(false, ah, false, bh, (short)0, acc, false, false);
    }

    // C/D 16x16 f32: VGPR r -> M = r + 8*half, N = lane&15
    const float bn = bias[n];
#pragma unroll
    for (int r = 0; r < 8; ++r) {
        const int mr = mbase + r + 8 * half;
        out[(size_t)mr * 256 + n] = leaky(acc[r] + bn);
    }
}

// ---------------------------------------------------------------------------
extern "C" void kernel_launch(void* const* d_in, const int* in_sizes, int n_in,
                              void* d_out, int out_size, void* d_ws, size_t ws_size,
                              hipStream_t stream) {
    const float* self_vecs = (const float*)d_in[0];   // (B,256)
    const float* neigh     = (const float*)d_in[1];   // (B,32,256)
    const float* W         = (const float*)d_in[2];   // (256,256)
    const float* attn      = (const float*)d_in[3];   // (256,1)
    const float* bias      = (const float*)d_in[4];   // (256)
    float* out = (float*)d_out;

    const int B = in_sizes[0] / 256;                  // 16384

    // Workspace layout (bytes):
    //   [0, 1024)                      wa        (256 f32)
    //   [1024, 132096)                 Wt_hi     (256*256 bf16)
    //   [132096, 263168)               Wt_lo
    //   [263168, 263168+8M)            s_hi      (B*256 bf16)
    //   [.. +8M, .. +16M)              s_lo
    char* ws = (char*)d_ws;
    float*  wa    = (float*)ws;
    __bf16* Wt_hi = (__bf16*)(ws + 1024);
    __bf16* Wt_lo = (__bf16*)(ws + 1024 + 131072);
    __bf16* s_hi  = (__bf16*)(ws + 263168);
    __bf16* s_lo  = (__bf16*)(ws + 263168 + (size_t)B * 256 * 2);

    prep_kernel<<<256, 256, 0, stream>>>(W, attn, wa, Wt_hi, Wt_lo);
    attn_kernel<<<B, 256, 0, stream>>>(self_vecs, neigh, wa, s_hi, s_lo);
    dim3 grid(256 / 16, B / 64);
    gemm_kernel<<<grid, 128, 0, stream>>>(s_hi, s_lo, Wt_hi, Wt_lo, bias, out);
}
